// GCN_En_27754078666885
// MI455X (gfx1250) — matrix operand order repaired and checked by reading, not compiled, not run on hardware
//
#include <hip/hip_runtime.h>

// GCN forward: out = relu(a @ (relu(a @ (x@W1) + b1) @ W2) + b2), a = adj/N
// N=16384, NFEAT=512, NHID=256, NEMBED=128. HBM-bound on streaming adj (2x1GiB).
// agg kernels: 8-wave WG, async-to-LDS double-buffered B panel, f16 WMMA, split-K=8.

#define N_NODES 16384
#define NFEAT   512
#define NHID    256
#define NEMBED  128
#define KSPLIT  8
#define KCHUNK  (N_NODES / KSPLIT)   // 2048
#define LDSROW  40                    // f16 elems per LDS panel row (80B: 16B-aligned, bank-spread)

typedef __attribute__((ext_vector_type(16))) _Float16 v16h;
typedef __attribute__((ext_vector_type(8)))  _Float16 v8h;
typedef __attribute__((ext_vector_type(8)))  float    v8f;
typedef __attribute__((ext_vector_type(4)))  float    v4f;

__device__ __forceinline__ v8f wmma16(v16h a, v16h b, v8f c) {
  return __builtin_amdgcn_wmma_f32_16x16x32_f16(false, a, false, b, (short)0, c, false, false);
}

__device__ __forceinline__ void wait_async0() {
#if __has_builtin(__builtin_amdgcn_s_wait_asynccnt)
  __builtin_amdgcn_s_wait_asynccnt(0);
#else
  asm volatile("s_wait_asynccnt 0x0" ::: "memory");
#endif
}

// One 16-byte async global->LDS copy (ASYNCcnt-tracked).
__device__ __forceinline__ void async_copy16(_Float16* lds_dst, const _Float16* gsrc) {
  unsigned loff = (unsigned)(uintptr_t)lds_dst;   // low 32 bits of generic LDS ptr = LDS offset
  asm volatile("global_load_async_to_lds_b128 %0, %1, off"
               :: "v"(loff), "v"(gsrc) : "memory");
}

// ---- A fragment (16x32 f16), layouts per CDNA5 ISA §7.12.2 ----
// Lane = 16*kh + m; holds K = k0+kh*8+{0..7} and k0+16+kh*8+{0..7}.
__device__ __forceinline__ void load_a_raw_nt(const float* __restrict__ base, int ld,
                                              int row0, int k0, int lane,
                                              v4f& f0, v4f& f1, v4f& f2, v4f& f3) {
  const int m = lane & 15, kh = lane >> 4;
  const float* p0 = base + (size_t)(row0 + m) * ld + (k0 + kh * 8);
  f0 = __builtin_nontemporal_load((const v4f*)(p0));
  f1 = __builtin_nontemporal_load((const v4f*)(p0 + 4));
  f2 = __builtin_nontemporal_load((const v4f*)(p0 + 16));
  f3 = __builtin_nontemporal_load((const v4f*)(p0 + 20));
}

__device__ __forceinline__ v16h cvt_a(v4f f0, v4f f1, v4f f2, v4f f3) {
  v16h r;
  r[0]  = (_Float16)f0.x; r[1]  = (_Float16)f0.y; r[2]  = (_Float16)f0.z; r[3]  = (_Float16)f0.w;
  r[4]  = (_Float16)f1.x; r[5]  = (_Float16)f1.y; r[6]  = (_Float16)f1.z; r[7]  = (_Float16)f1.w;
  r[8]  = (_Float16)f2.x; r[9]  = (_Float16)f2.y; r[10] = (_Float16)f2.z; r[11] = (_Float16)f2.w;
  r[12] = (_Float16)f3.x; r[13] = (_Float16)f3.y; r[14] = (_Float16)f3.z; r[15] = (_Float16)f3.w;
  return r;
}

__device__ __forceinline__ v16h load_a_f32(const float* __restrict__ base, int ld,
                                           int row0, int k0, int lane) {
  const int m = lane & 15, kh = lane >> 4;
  const float* p0 = base + (size_t)(row0 + m) * ld + (k0 + kh * 8);
  v4f f0 = *(const v4f*)(p0);
  v4f f1 = *(const v4f*)(p0 + 4);
  v4f f2 = *(const v4f*)(p0 + 16);
  v4f f3 = *(const v4f*)(p0 + 20);
  return cvt_a(f0, f1, f2, f3);
}

__device__ __forceinline__ v16h load_a_f16(const _Float16* __restrict__ base, int ld,
                                           int row0, int k0, int lane) {
  const int m = lane & 15, kh = lane >> 4;
  const _Float16* p0 = base + (size_t)(row0 + m) * ld + (k0 + kh * 8);
  v8h lo = *(const v8h*)(p0);
  v8h hi = *(const v8h*)(p0 + 16);
  return __builtin_shufflevector(lo, hi, 0,1,2,3,4,5,6,7,8,9,10,11,12,13,14,15);
}

// B fragment 32x16 from transposed f16 global storage bt[col][k], tile at (k0, col0).
// Lane = 16*kh + n; holds K = k0+kh*8+{0..7} and k0+16+kh*8+{0..7}.
__device__ __forceinline__ v16h load_b_t(const _Float16* __restrict__ bt, int ld,
                                         int k0, int col0, int lane) {
  const int n = lane & 15, kh = lane >> 4;
  const _Float16* p = bt + (size_t)(col0 + n) * ld + (k0 + kh * 8);
  v8h lo = *(const v8h*)(p);
  v8h hi = *(const v8h*)(p + 16);
  return __builtin_shufflevector(lo, hi, 0,1,2,3,4,5,6,7,8,9,10,11,12,13,14,15);
}

// B fragment from LDS panel: panel[col][0..31] holds K=kk..kk+31 for that column.
__device__ __forceinline__ v16h ldsb(const _Float16 (*panel)[LDSROW], int t, int lane) {
  const int n = lane & 15, kh = lane >> 4;
  const _Float16* p = &panel[t * 16 + n][kh * 8];
  v8h lo = *(const v8h*)(p);
  v8h hi = *(const v8h*)(p + 16);
  return __builtin_shufflevector(lo, hi, 0,1,2,3,4,5,6,7,8,9,10,11,12,13,14,15);
}

// Async fill of one panel: NCOLS rows x 64B (K slab [kk,kk+32)), one row per thread.
template <int NCOLS>
__device__ __forceinline__ void fill_async(_Float16 (*dst)[LDSROW],
                                           const _Float16* __restrict__ bt, int kk, int tid) {
  if (tid < NCOLS) {
    const _Float16* g = bt + (size_t)tid * N_NODES + kk;
    _Float16* l = &dst[tid][0];
    async_copy16(l,      g);
    async_copy16(l + 8,  g + 8);
    async_copy16(l + 16, g + 16);
    async_copy16(l + 24, g + 24);
  }
}

// ---- Prep: W [K][Ncol] fp32 -> Wt [Ncol][K] f16 ----
__global__ void convert_transpose_kernel(const float* __restrict__ w, _Float16* __restrict__ wt,
                                         int K, int Ncol) {
  int idx = blockIdx.x * blockDim.x + threadIdx.x;
  if (idx >= K * Ncol) return;
  int k = idx / Ncol, n = idx - k * Ncol;
  wt[(size_t)n * K + k] = (_Float16)w[idx];
}

// ---- GEMM1: P1t = (x @ W1)^T f16 [NHID][N_NODES]. One wave per 16 rows. ----
__global__ void __launch_bounds__(32)
gemm1_kernel(const float* __restrict__ x, const _Float16* __restrict__ w1t,
             _Float16* __restrict__ p1t) {
  const int lane = threadIdx.x;
  const int r0 = blockIdx.x * 16;
  v8f acc[NHID / 16] = {};
  for (int kk = 0; kk < NFEAT; kk += 32) {
    v16h a = load_a_f32(x, NFEAT, r0, kk, lane);
    #pragma unroll
    for (int t = 0; t < NHID / 16; ++t) {
      v16h b = load_b_t(w1t, NFEAT, kk, t * 16, lane);
      acc[t] = wmma16(a, b, acc[t]);
    }
  }
  const int n = lane & 15, kh = lane >> 4;
  #pragma unroll
  for (int t = 0; t < NHID / 16; ++t)
    #pragma unroll
    for (int i = 0; i < 8; ++i)
      p1t[(size_t)(t * 16 + n) * N_NODES + (r0 + i + 8 * kh)] = (_Float16)acc[t][i];
}

// ---- AGG: partial[s] = adj[128 rows, Kchunk_s] @ B. 8 waves share async-LDS B panel. ----
template <int NCOLS>
__global__ void __launch_bounds__(256)
agg_kernel(const float* __restrict__ adj, const _Float16* __restrict__ bt,
           float* __restrict__ partial) {
  constexpr int NT = NCOLS / 16;
  __shared__ _Float16 bpanel[2][NCOLS][LDSROW];

  const int tid  = threadIdx.x;
  const int lane = tid & 31;
  const int wave = tid >> 5;
  const int r0    = blockIdx.x * 128 + wave * 16;
  const int kbase = blockIdx.y * KCHUNK;
  const int kend  = kbase + KCHUNK;

  v8f acc[NT] = {};

  // Preload: A regs for first K-step, async-fill LDS buffer 0.
  v4f c0, c1, c2, c3, n0, n1, n2, n3;
  load_a_raw_nt(adj, N_NODES, r0, kbase, lane, c0, c1, c2, c3);
  fill_async<NCOLS>(bpanel[0], bt, kbase, tid);

  int cur = 0;
  for (int kk = kbase; kk < kend; kk += 32, cur ^= 1) {
    const int kn = kk + 32;
    const bool more = (kn < kend);
    if (more) load_a_raw_nt(adj, N_NODES, r0, kn, lane, n0, n1, n2, n3);  // next A in flight

    wait_async0();        // my fill of bpanel[cur] is complete
    __syncthreads();      // everyone's fill complete; everyone done reading bpanel[cur^1]
    if (more) fill_async<NCOLS>(bpanel[cur ^ 1], bt, kn, tid);

    v16h a = cvt_a(c0, c1, c2, c3);
    // 2-deep register pipeline over the LDS B fragments.
    v16h b0 = ldsb(bpanel[cur], 0, lane);
    v16h b1 = ldsb(bpanel[cur], 1, lane);
    #pragma unroll
    for (int t = 0; t < NT; ++t) {
      v16h bn = (t + 2 < NT) ? ldsb(bpanel[cur], t + 2, lane) : b0;
      acc[t] = wmma16(a, (t & 1) ? b1 : b0, acc[t]);
      if (t & 1) b1 = bn; else b0 = bn;
    }
    if (more) { c0 = n0; c1 = n1; c2 = n2; c3 = n3; }
  }

  float* pp = partial + (size_t)blockIdx.y * N_NODES * NCOLS;
  const int n = lane & 15, kh = lane >> 4;
  #pragma unroll
  for (int t = 0; t < NT; ++t)
    #pragma unroll
    for (int i = 0; i < 8; ++i)
      pp[(size_t)(r0 + i + 8 * kh) * NCOLS + (t * 16 + n)] = acc[t][i];
}

// ---- Epilogue: deterministic split-K reduce + relu(sum/N + bias). ----
template <int NCOLS, bool OUTF16>
__global__ void epilogue_kernel(const float* __restrict__ part, const float* __restrict__ bias,
                                void* __restrict__ outp, float scale) {
  const size_t total = (size_t)N_NODES * NCOLS;
  size_t idx = (size_t)blockIdx.x * blockDim.x + threadIdx.x;
  if (idx >= total) return;
  float s = 0.f;
  #pragma unroll
  for (int p = 0; p < KSPLIT; ++p) s += part[(size_t)p * total + idx];
  s = s * scale + bias[idx & (NCOLS - 1)];
  s = s > 0.f ? s : 0.f;
  if (OUTF16) ((_Float16*)outp)[idx] = (_Float16)s;
  else        ((float*)outp)[idx] = s;
}

// ---- GEMM2: P2t = (H @ W2)^T f16 [NEMBED][N_NODES]. ----
__global__ void __launch_bounds__(32)
gemm2_kernel(const _Float16* __restrict__ h, const _Float16* __restrict__ w2t,
             _Float16* __restrict__ p2t) {
  const int lane = threadIdx.x;
  const int r0 = blockIdx.x * 16;
  v8f acc[NEMBED / 16] = {};
  for (int kk = 0; kk < NHID; kk += 32) {
    v16h a = load_a_f16(h, NHID, r0, kk, lane);
    #pragma unroll
    for (int t = 0; t < NEMBED / 16; ++t) {
      v16h b = load_b_t(w2t, NHID, kk, t * 16, lane);
      acc[t] = wmma16(a, b, acc[t]);
    }
  }
  const int n = lane & 15, kh = lane >> 4;
  #pragma unroll
  for (int t = 0; t < NEMBED / 16; ++t)
    #pragma unroll
    for (int i = 0; i < 8; ++i)
      p2t[(size_t)(t * 16 + n) * N_NODES + (r0 + i + 8 * kh)] = (_Float16)acc[t][i];
}

extern "C" void kernel_launch(void* const* d_in, const int* in_sizes, int n_in,
                              void* d_out, int out_size, void* d_ws, size_t ws_size,
                              hipStream_t stream) {
  const float* x   = (const float*)d_in[0];
  const float* adj = (const float*)d_in[1];
  const float* W1  = (const float*)d_in[2];
  const float* b1  = (const float*)d_in[3];
  const float* W2  = (const float*)d_in[4];
  const float* b2  = (const float*)d_in[5];
  float* out = (float*)d_out;

  char* ws = (char*)d_ws;
  _Float16* w1t  = (_Float16*)(ws);                               // 256 KiB
  _Float16* w2t  = (_Float16*)(ws + (256u << 10));                // 64 KiB
  _Float16* p1t  = (_Float16*)(ws + (512u << 10));                // 8 MiB  [NHID][N]
  _Float16* hbuf = (_Float16*)(ws + (512u << 10) + (8u << 20));   // 8 MiB  [N][NHID]
  _Float16* p2t  = (_Float16*)(ws + (512u << 10) + (16u << 20));  // 4 MiB  [NEMBED][N]
  float* partial = (float*)(ws + (512u << 10) + (20u << 20));     // <=128 MiB, reused

  const float scale = 1.0f / (float)N_NODES;

  convert_transpose_kernel<<<(NFEAT * NHID + 255) / 256, 256, 0, stream>>>(W1, w1t, NFEAT, NHID);
  convert_transpose_kernel<<<(NHID * NEMBED + 255) / 256, 256, 0, stream>>>(W2, w2t, NHID, NEMBED);

  // Layer 1
  gemm1_kernel<<<N_NODES / 16, 32, 0, stream>>>(x, w1t, p1t);
  agg_kernel<NHID><<<dim3(N_NODES / 128, KSPLIT), 256, 0, stream>>>(adj, p1t, partial);
  epilogue_kernel<NHID, true><<<(N_NODES * NHID + 255) / 256, 256, 0, stream>>>(
      partial, b1, (void*)hbuf, scale);

  // Layer 2
  gemm2_kernel<<<N_NODES / 16, 32, 0, stream>>>(hbuf, w2t, p2t);
  agg_kernel<NEMBED><<<dim3(N_NODES / 128, KSPLIT), 256, 0, stream>>>(adj, p2t, partial);
  epilogue_kernel<NEMBED, false><<<(N_NODES * NEMBED + 255) / 256, 256, 0, stream>>>(
      partial, b2, (void*)out, scale);
}
